// DeepARNetwork_14714557956432
// MI455X (gfx1250) — matrix operand order repaired and checked by reading, not compiled, not run on hardware
//
#include <hip/hip_runtime.h>

#define BATCH 8192
#define CTXL  168
#define SPL   24
#define HISTL 336
#define NCH   128
#define NSEGS 7

typedef __attribute__((ext_vector_type(16))) _Float16 v16h;
typedef __attribute__((ext_vector_type(8)))  _Float16 v8h;
typedef __attribute__((ext_vector_type(8)))  float    v8f;

__device__ __forceinline__ v8f wmma_fp16(v16h a, v16h b, v8f c) {
    return __builtin_amdgcn_wmma_f32_16x16x32_f16(false, a, false, b, (short)0, c,
                                                  false, false);
}

// A fragment (16x32 f16, MxK): lane L holds row M=(L&15); lanes 0-15 carry
// K {k0..k0+7, k0+16..k0+23}, lanes 16-31 carry K {k0+8..k0+15, k0+24..k0+31}.
__device__ __forceinline__ v16h load_a_frag(const _Float16* rowptr, int k0, int laneHi) {
    const int o = laneHi ? 8 : 0;
    v8h lo = *(const v8h*)(rowptr + k0 + o);
    v8h hi = *(const v8h*)(rowptr + k0 + 16 + o);
    v16h r;
#pragma unroll
    for (int i = 0; i < 8; ++i) { r[i] = lo[i]; r[8 + i] = hi[i]; }
    return r;
}

// B fragment (32x16 f16, KxN): column n = lane&15 == weight row (nrow);
// lanes 0-15 hold K {k0..k0+7, k0+16..k0+23}, lanes 16-31 K {k0+8..k0+15, k0+24..k0+31}.
// Weights stored [N, K] row-major, so column n of B == row n of W.
__device__ __forceinline__ v16h load_b_frag(const _Float16* __restrict__ W, int ldw,
                                            int nrow, int k0, int laneHi) {
    const _Float16* p = W + (size_t)nrow * ldw + k0 + (laneHi ? 8 : 0);
    v8h lo = *(const v8h*)(p);
    v8h hi = *(const v8h*)(p + 16);
    v16h r;
#pragma unroll
    for (int i = 0; i < 8; ++i) { r[i] = lo[i]; r[8 + i] = hi[i]; }
    return r;
}

__device__ __forceinline__ float sigmoidf_(float x) { return 1.0f / (1.0f + __expf(-x)); }

__device__ const int LAGS_d[8] = {24, 48, 72, 96, 120, 144, 168, 0};

// ---------------- helper kernels ----------------

__global__ void cvt_pad(const float* __restrict__ src, _Float16* __restrict__ dst,
                        int N, int K, int Kp) {
    long i = (long)blockIdx.x * 256 + threadIdx.x;
    long tot = (long)N * Kp;
    if (i >= tot) return;
    int n = (int)(i / Kp), k = (int)(i % Kp);
    dst[i] = (_Float16)(k < K ? src[(long)n * K + k] : 0.0f);
}

__global__ void zero_h16(_Float16* p, long n) {
    long i = (long)blockIdx.x * 256 + threadIdx.x;
    if (i < n) p[i] = (_Float16)0.0f;
}
__global__ void zero_f32(float* p, long n) {
    long i = (long)blockIdx.x * 256 + threadIdx.x;
    if (i < n) p[i] = 0.0f;
}

// scale: per-row |x| sum over the last CTX targets (observed weights are all 1)
__global__ void scale1(const float* __restrict__ ptg, float* __restrict__ wsum) {
    __shared__ float sm[256];
    int b = blockIdx.x, tid = threadIdx.x;
    float v = 0.0f;
    if (tid < CTXL) v = fabsf(ptg[(size_t)b * HISTL + CTXL + tid]);
    sm[tid] = v; __syncthreads();
    for (int s = 128; s > 0; s >>= 1) { if (tid < s) sm[tid] += sm[tid + s]; __syncthreads(); }
    if (tid == 0) wsum[b] = sm[0];
}
__global__ void scale2(const float* __restrict__ wsum, float* __restrict__ dscale) {
    __shared__ float sm[256];
    float v = 0.0f;
    for (int i = threadIdx.x; i < BATCH; i += 256) v += wsum[i];
    sm[threadIdx.x] = v; __syncthreads();
    for (int s = 128; s > 0; s >>= 1) { if (threadIdx.x < s) sm[threadIdx.x] += sm[threadIdx.x + s]; __syncthreads(); }
    if (threadIdx.x == 0) dscale[0] = sm[0] / (float)((long)CTXL * BATCH);
}
__global__ void scale3(const float* __restrict__ wsum, const float* __restrict__ dscale,
                       float* __restrict__ scl, float* __restrict__ lscl) {
    int b = blockIdx.x * 256 + threadIdx.x;
    if (b >= BATCH) return;
    float ws = wsum[b];
    float s = ws > 0.0f ? ws / (float)CTXL : dscale[0];
    s = fmaxf(1e-10f, s);
    scl[b] = s;
    lscl[b] = __logf(s);
}

// Build X f16 [BATCH, CTX, 32]: 8 scaled lags | 4 time feats | 16 emb | real | log(scale) | pad
__global__ void buildX(const float* __restrict__ ptg, const float* __restrict__ ptf,
                       const int* __restrict__ cat, const float* __restrict__ sreal,
                       const float* __restrict__ emb, const float* __restrict__ scl,
                       const float* __restrict__ lscl, _Float16* __restrict__ X) {
    int idx = blockIdx.x * 8 + (threadIdx.x >> 5);   // row over BATCH*CTX
    int col = threadIdx.x & 31;
    int b = idx / CTXL, s = idx % CTXL;
    float v;
    if (col < 8) {
        int lag = LAGS_d[col];
        v = ptg[(size_t)b * HISTL + (CTXL - lag + s)] / scl[b];
    } else if (col < 12) {
        v = ptf[((size_t)b * HISTL + CTXL + s) * 4 + (col - 8)];
    } else if (col < 28) {
        v = emb[cat[b] * 16 + (col - 12)];
    } else if (col == 28) v = sreal[b];
    else if (col == 29)   v = lscl[b];
    else                  v = 0.0f;
    X[(size_t)idx * 32 + col] = (_Float16)v;
}

// Build decoder query f16 [BATCH*SP, 192]
__global__ void buildQ(const _Float16* __restrict__ X, const _Float16* __restrict__ enc,
                       const float* __restrict__ ftf, const int* __restrict__ cat,
                       const float* __restrict__ sreal, const float* __restrict__ emb,
                       const float* __restrict__ lscl, _Float16* __restrict__ Q) {
    int row = blockIdx.x;                 // b*SP + t
    int col = threadIdx.x;                // 0..191
    int b = row / SPL, t = row % SPL;
    _Float16 v;
    if (col < 30)        v = X[((size_t)b * CTXL + t) * 32 + col];
    else if (col < 158)  v = enc[(size_t)row * NCH + (col - 30)];
    else if (col < 162)  v = (_Float16)ftf[((size_t)b * SPL + t) * 4 + (col - 158)];
    else if (col < 178)  v = (_Float16)emb[cat[b] * 16 + (col - 162)];
    else if (col == 178) v = (_Float16)sreal[b];
    else if (col == 179) v = (_Float16)lscl[b];
    else                 v = (_Float16)0.0f;
    Q[(size_t)row * 192 + col] = v;
}

// ---------------- WMMA GRU: hv = GRU(x, h_prev_encoder_out) ----------------
// MT=2 M-tiles per block (32 rows). A tile staged in LDS (shared by 8 waves),
// B fragments reused across both M-tiles. grid = BATCH*SP/32, 256 threads.
__global__ __launch_bounds__(256) void gru_kernel(
    const _Float16* __restrict__ X,    // [BATCH*CTX, 32]
    const _Float16* __restrict__ Hin,  // [BATCH*SP, 128]
    const _Float16* __restrict__ Wih,  // [384, 32] f16
    const _Float16* __restrict__ Whh,  // [384, 128] f16
    const float* __restrict__ bih, const float* __restrict__ bhh,
    _Float16* __restrict__ Hv, int seg) {
    __shared__ __align__(16) _Float16 xs[32 * 32];
    __shared__ __align__(16) _Float16 hs[32 * NCH];

    const int tid = threadIdx.x;
    const int lane = tid & 31, wave = tid >> 5;
    const int laneHi = lane >> 4, lrow = lane & 15;
    const int tileM0 = blockIdx.x * 32;

    // cooperative staging: x chunks 32*4=128, h chunks 32*16=512 (v8h each)
    for (int idx = tid; idx < 640; idx += 256) {
        if (idx < 128) {
            int r = idx >> 2, c = idx & 3;
            int m = tileM0 + r, b = m / SPL, t = m % SPL;
            *(v8h*)(xs + r * 32 + c * 8) =
                *(const v8h*)(X + ((size_t)b * CTXL + seg * SPL + t) * 32 + c * 8);
        } else {
            int j = idx - 128;
            int r = j >> 4, c = j & 15;
            *(v8h*)(hs + r * NCH + c * 8) =
                *(const v8h*)(Hin + (size_t)(tileM0 + r) * NCH + c * 8);
        }
    }
    __syncthreads();

    const int nb = wave * 16 + lrow;      // weight row within each gate block
    v8f accR[2] = {{}, {}}, accZ[2] = {{}, {}}, accI[2] = {{}, {}}, accH[2] = {{}, {}};

    {   // x part: K = 32 (30 padded)
        v16h bR = load_b_frag(Wih, 32, 0 * NCH + nb, 0, laneHi);
        v16h bZ = load_b_frag(Wih, 32, 1 * NCH + nb, 0, laneHi);
        v16h bI = load_b_frag(Wih, 32, 2 * NCH + nb, 0, laneHi);
#pragma unroll
        for (int mt = 0; mt < 2; ++mt) {
            v16h a = load_a_frag(xs + (mt * 16 + lrow) * 32, 0, laneHi);
            accR[mt] = wmma_fp16(a, bR, accR[mt]);
            accZ[mt] = wmma_fp16(a, bZ, accZ[mt]);
            accI[mt] = wmma_fp16(a, bI, accI[mt]);
        }
    }
#pragma unroll
    for (int k0 = 0; k0 < NCH; k0 += 32) {   // h part: K = 128
        v16h bR = load_b_frag(Whh, NCH, 0 * NCH + nb, k0, laneHi);
        v16h bZ = load_b_frag(Whh, NCH, 1 * NCH + nb, k0, laneHi);
        v16h bH = load_b_frag(Whh, NCH, 2 * NCH + nb, k0, laneHi);
#pragma unroll
        for (int mt = 0; mt < 2; ++mt) {
            v16h a = load_a_frag(hs + (mt * 16 + lrow) * NCH, k0, laneHi);
            accR[mt] = wmma_fp16(a, bR, accR[mt]);
            accZ[mt] = wmma_fp16(a, bZ, accZ[mt]);
            accH[mt] = wmma_fp16(a, bH, accH[mt]);
        }
    }

    const int col = nb;
    const float brv  = bih[col] + bhh[col];
    const float bzv  = bih[NCH + col] + bhh[NCH + col];
    const float binv = bih[2 * NCH + col];
    const float bhnv = bhh[2 * NCH + col];
#pragma unroll
    for (int mt = 0; mt < 2; ++mt) {
#pragma unroll
        for (int i = 0; i < 8; ++i) {   // C layout: VGPR i -> row +i+8*laneHi, col = lane&15
            const int rl = mt * 16 + i + laneHi * 8;
            float hval = (float)hs[rl * NCH + col];
            float r = sigmoidf_(accR[mt][i] + brv);
            float z = sigmoidf_(accZ[mt][i] + bzv);
            float n = tanhf(accI[mt][i] + binv + r * (accH[mt][i] + bhnv));
            Hv[(size_t)(tileM0 + rl) * NCH + col] = (_Float16)((1.0f - z) * n + z * hval);
        }
    }
}

// ---------------- WMMA LSTM step (encoder & decoder) ----------------
// gates = x_t @ Wih^T + h @ Whh^T + b, fused; MT=2; grid = BATCH/32, 256 threads.
// H is staged to LDS before the barrier, so the later global H writes are safe.
template <int K1>
__global__ __launch_bounds__(256) void lstm_step(
    const _Float16* __restrict__ Xin,   // [BATCH*SP, K1], row (b*SP + t)
    const _Float16* __restrict__ Wih,   // [512, K1] f16
    const _Float16* __restrict__ Whh,   // [512, 128] f16
    const float* __restrict__ bias,     // [512]
    _Float16* __restrict__ H,           // [BATCH, 128] f16
    float* __restrict__ Cst,            // [BATCH, 128] f32
    _Float16* __restrict__ Yout,        // [BATCH*SP, 128] f16
    int t) {
    __shared__ __align__(16) _Float16 xs[32 * K1];
    __shared__ __align__(16) _Float16 hs[32 * NCH];

    const int tid = threadIdx.x;
    const int lane = tid & 31, wave = tid >> 5;
    const int laneHi = lane >> 4, lrow = lane & 15;
    const int tileM0 = blockIdx.x * 32;

    const int XCH = 32 * (K1 / 8);              // 512 (K1=128) or 768 (K1=192)
    for (int idx = tid; idx < XCH + 512; idx += 256) {
        if (idx < XCH) {
            int r = idx / (K1 / 8), c = idx % (K1 / 8);
            *(v8h*)(xs + r * K1 + c * 8) =
                *(const v8h*)(Xin + ((size_t)(tileM0 + r) * SPL + t) * K1 + c * 8);
        } else {
            int j = idx - XCH;
            int r = j >> 4, c = j & 15;
            *(v8h*)(hs + r * NCH + c * 8) =
                *(const v8h*)(H + (size_t)(tileM0 + r) * NCH + c * 8);
        }
    }
    __syncthreads();

    const int nb = wave * 16 + lrow;
    v8f acc[2][4] = {{{}, {}, {}, {}}, {{}, {}, {}, {}}};

#pragma unroll
    for (int k0 = 0; k0 < K1; k0 += 32) {
        v16h b0 = load_b_frag(Wih, K1, 0 * NCH + nb, k0, laneHi);
        v16h b1 = load_b_frag(Wih, K1, 1 * NCH + nb, k0, laneHi);
        v16h b2 = load_b_frag(Wih, K1, 2 * NCH + nb, k0, laneHi);
        v16h b3 = load_b_frag(Wih, K1, 3 * NCH + nb, k0, laneHi);
#pragma unroll
        for (int mt = 0; mt < 2; ++mt) {
            v16h a = load_a_frag(xs + (mt * 16 + lrow) * K1, k0, laneHi);
            acc[mt][0] = wmma_fp16(a, b0, acc[mt][0]);
            acc[mt][1] = wmma_fp16(a, b1, acc[mt][1]);
            acc[mt][2] = wmma_fp16(a, b2, acc[mt][2]);
            acc[mt][3] = wmma_fp16(a, b3, acc[mt][3]);
        }
    }
#pragma unroll
    for (int k0 = 0; k0 < NCH; k0 += 32) {
        v16h b0 = load_b_frag(Whh, NCH, 0 * NCH + nb, k0, laneHi);
        v16h b1 = load_b_frag(Whh, NCH, 1 * NCH + nb, k0, laneHi);
        v16h b2 = load_b_frag(Whh, NCH, 2 * NCH + nb, k0, laneHi);
        v16h b3 = load_b_frag(Whh, NCH, 3 * NCH + nb, k0, laneHi);
#pragma unroll
        for (int mt = 0; mt < 2; ++mt) {
            v16h a = load_a_frag(hs + (mt * 16 + lrow) * NCH, k0, laneHi);
            acc[mt][0] = wmma_fp16(a, b0, acc[mt][0]);
            acc[mt][1] = wmma_fp16(a, b1, acc[mt][1]);
            acc[mt][2] = wmma_fp16(a, b2, acc[mt][2]);
            acc[mt][3] = wmma_fp16(a, b3, acc[mt][3]);
        }
    }

    const int col = nb;
    const float bi = bias[col], bf = bias[NCH + col];
    const float bg = bias[2 * NCH + col], bo = bias[3 * NCH + col];
#pragma unroll
    for (int mt = 0; mt < 2; ++mt) {
#pragma unroll
        for (int i = 0; i < 8; ++i) {
            const int row = tileM0 + mt * 16 + i + laneHi * 8;
            const size_t ho = (size_t)row * NCH + col;
            float c_old = Cst[ho];
            float ig = sigmoidf_(acc[mt][0][i] + bi);
            float fg = sigmoidf_(acc[mt][1][i] + bf);
            float gg = tanhf(acc[mt][2][i] + bg);
            float og = sigmoidf_(acc[mt][3][i] + bo);
            float cn = fg * c_old + ig * gg;
            float hn = og * tanhf(cn);
            Cst[ho] = cn;
            H[ho] = (_Float16)hn;
            Yout[((size_t)row * SPL + t) * NCH + col] = (_Float16)hn;
        }
    }
}

// out[b,t] = (dec_out[b,t,:] . out_W + out_b) * scale[b]; one wave per row
__global__ __launch_bounds__(256) void out_kernel(
    const _Float16* __restrict__ dec, const float* __restrict__ oW,
    const float* __restrict__ ob, const float* __restrict__ scl,
    float* __restrict__ out) {
    int row = blockIdx.x * 8 + (threadIdx.x >> 5);
    int lane = threadIdx.x & 31;
    const _Float16* p = dec + (size_t)row * NCH;
    float s = 0.0f;
#pragma unroll
    for (int k = lane; k < NCH; k += 32) s += (float)p[k] * oW[k];
#pragma unroll
    for (int off = 16; off > 0; off >>= 1) s += __shfl_down(s, off, 32);
    if (lane == 0) out[row] = (s + ob[0]) * scl[row / SPL];
}

// ---------------- launch ----------------

extern "C" void kernel_launch(void* const* d_in, const int* in_sizes, int n_in,
                              void* d_out, int out_size, void* d_ws, size_t ws_size,
                              hipStream_t stream) {
    (void)in_sizes; (void)n_in; (void)out_size; (void)ws_size;
    const int*   cat   = (const int*)d_in[0];
    const float* sreal = (const float*)d_in[1];
    const float* ptf   = (const float*)d_in[2];
    const float* ptg   = (const float*)d_in[3];
    const float* ftf   = (const float*)d_in[5];
    const float* emb   = (const float*)d_in[6];
    const float* gWih  = (const float*)d_in[7];
    const float* gWhh  = (const float*)d_in[8];
    const float* gbih  = (const float*)d_in[9];
    const float* gbhh  = (const float*)d_in[10];
    const float* eWih  = (const float*)d_in[11];
    const float* eWhh  = (const float*)d_in[12];
    const float* eb    = (const float*)d_in[13];
    const float* dWih  = (const float*)d_in[14];
    const float* dWhh  = (const float*)d_in[15];
    const float* db    = (const float*)d_in[16];
    const float* oW    = (const float*)d_in[17];
    const float* ob    = (const float*)d_in[18];
    float* out = (float*)d_out;

    char* ws = (char*)d_ws;
    size_t off = 0;
    auto alloc = [&](size_t bytes) -> char* {
        char* p = ws + off; off += (bytes + 255) & ~(size_t)255; return p;
    };
    _Float16* X      = (_Float16*)alloc((size_t)BATCH * CTXL * 32 * 2);
    _Float16* Enc    = (_Float16*)alloc((size_t)BATCH * SPL * NCH * 2);
    _Float16* Hv     = (_Float16*)alloc((size_t)BATCH * SPL * NCH * 2);  // also dec_out
    _Float16* Q      = (_Float16*)alloc((size_t)BATCH * SPL * 192 * 2);
    _Float16* H      = (_Float16*)alloc((size_t)BATCH * NCH * 2);
    float*    Cst    = (float*)alloc((size_t)BATCH * NCH * 4);
    float*    wsum   = (float*)alloc(BATCH * 4);
    float*    scl    = (float*)alloc(BATCH * 4);
    float*    lscl   = (float*)alloc(BATCH * 4);
    float*    dscale = (float*)alloc(256);
    _Float16* gWih16 = (_Float16*)alloc(384 * 32 * 2);
    _Float16* gWhh16 = (_Float16*)alloc(384 * 128 * 2);
    _Float16* eWih16 = (_Float16*)alloc((size_t)NSEGS * 512 * 128 * 2);
    _Float16* eWhh16 = (_Float16*)alloc((size_t)NSEGS * 512 * 128 * 2);
    _Float16* dWih16 = (_Float16*)alloc(512 * 192 * 2);
    _Float16* dWhh16 = (_Float16*)alloc(512 * 128 * 2);

    auto blk = [](long n) { return (unsigned)((n + 255) / 256); };

    // weight conversion to padded f16
    cvt_pad<<<blk(384L * 32), 256, 0, stream>>>(gWih, gWih16, 384, 30, 32);
    cvt_pad<<<blk(384L * 128), 256, 0, stream>>>(gWhh, gWhh16, 384, 128, 128);
    cvt_pad<<<blk((long)NSEGS * 512 * 128), 256, 0, stream>>>(eWih, eWih16, NSEGS * 512, 128, 128);
    cvt_pad<<<blk((long)NSEGS * 512 * 128), 256, 0, stream>>>(eWhh, eWhh16, NSEGS * 512, 128, 128);
    cvt_pad<<<blk(512L * 192), 256, 0, stream>>>(dWih, dWih16, 512, 180, 192);
    cvt_pad<<<blk(512L * 128), 256, 0, stream>>>(dWhh, dWhh16, 512, 128, 128);

    // state init (deterministic each call)
    zero_h16<<<blk((long)BATCH * SPL * NCH), 256, 0, stream>>>(Enc, (long)BATCH * SPL * NCH);
    zero_h16<<<blk((long)BATCH * NCH), 256, 0, stream>>>(H, (long)BATCH * NCH);
    zero_f32<<<blk((long)BATCH * NCH), 256, 0, stream>>>(Cst, (long)BATCH * NCH);

    // scale
    scale1<<<BATCH, 256, 0, stream>>>(ptg, wsum);
    scale2<<<1, 256, 0, stream>>>(wsum, dscale);
    scale3<<<BATCH / 256, 256, 0, stream>>>(wsum, dscale, scl, lscl);

    // input tensor
    buildX<<<(BATCH * CTXL) / 8, 256, 0, stream>>>(ptg, ptf, cat, sreal, emb, scl, lscl, X);

    // encoder: per segment, GRU then 24 LSTM steps
    for (int seg = 0; seg < NSEGS; ++seg) {
        gru_kernel<<<(BATCH * SPL) / 32, 256, 0, stream>>>(X, Enc, gWih16, gWhh16,
                                                           gbih, gbhh, Hv, seg);
        const _Float16* wih = eWih16 + (size_t)seg * 512 * 128;
        const _Float16* whh = eWhh16 + (size_t)seg * 512 * 128;
        const float*    bb  = eb + (size_t)seg * 512;
        for (int t = 0; t < SPL; ++t)
            lstm_step<128><<<BATCH / 32, 256, 0, stream>>>(Hv, wih, whh, bb, H, Cst, Enc, t);
    }

    // decoder
    buildQ<<<BATCH * SPL, 192, 0, stream>>>(X, Enc, ftf, cat, sreal, emb, lscl, Q);
    for (int t = 0; t < SPL; ++t)
        lstm_step<192><<<BATCH / 32, 256, 0, stream>>>(Q, dWih16, dWhh16, db, H, Cst, Hv, t);

    // projection + rescale
    out_kernel<<<(BATCH * SPL) / 8, 256, 0, stream>>>(Hv, oW, ob, scl, out);
}
